// FlashAttention_15857019256895
// MI455X (gfx1250) — compile-verified, matrix operand order checked by reading, not run on hardware
//
#include <hip/hip_runtime.h>

typedef __attribute__((ext_vector_type(16))) _Float16 v16h;
typedef __attribute__((ext_vector_type(8)))  float    v8f;

constexpr int B_ = 4, T_ = 2048, C_ = 2048, H_ = 16, D_ = 128;
constexpr int QKV_F = 3 * C_;                 // 6144
constexpr float SCALE = 0.08838834764831845f; // 1/sqrt(128)

// K-index for 16-bit A/B fragment slot v (each slot = 2 packed halves),
// per CDNA5 ISA 7.12.2: lanes 0-15 -> K 0..7 / 16..23, lanes 16-31 -> K 8..15 / 24..31
__device__ __forceinline__ int kslot(int v, int half) {
  return (v < 4 ? 2 * v : 16 + 2 * (v - 4)) + half * 8;
}

__device__ __forceinline__ v16h load_afrag(const _Float16* base, int ld, int lane) {
  const int m = lane & 15, half = lane >> 4;
  v16h a;
#pragma unroll
  for (int v = 0; v < 8; ++v) {
    const int k = kslot(v, half);
    a[2 * v]     = base[m * ld + k];
    a[2 * v + 1] = base[m * ld + k + 1];
  }
  return a;
}

// B fragment when data is stored [n][k]
__device__ __forceinline__ v16h load_bfrag_nk(const _Float16* base, int ld, int lane) {
  const int n = lane & 15, half = lane >> 4;
  v16h b;
#pragma unroll
  for (int v = 0; v < 8; ++v) {
    const int k = kslot(v, half);
    b[2 * v]     = base[n * ld + k];
    b[2 * v + 1] = base[n * ld + k + 1];
  }
  return b;
}

// B fragment when data is stored [k][n]
__device__ __forceinline__ v16h load_bfrag_kn(const _Float16* base, int ld, int lane) {
  const int n = lane & 15, half = lane >> 4;
  v16h b;
#pragma unroll
  for (int v = 0; v < 8; ++v) {
    const int k = kslot(v, half);
    b[2 * v]     = base[k * ld + n];
    b[2 * v + 1] = base[(k + 1) * ld + n];
  }
  return b;
}

__device__ __forceinline__ v8f wmma16(v16h a, v16h b, v8f c) {
  return __builtin_amdgcn_wmma_f32_16x16x32_f16(false, a, false, b, (short)0, c,
                                                false, false);
}

// One async 16-byte global->LDS copy (ASYNCcnt path).
__device__ __forceinline__ void async_copy_b128(unsigned lds, unsigned voff,
                                                unsigned long long sbase) {
  asm volatile("global_load_async_to_lds_b128 %0, %1, %2"
               :: "v"(lds), "v"(voff), "s"(sbase) : "memory");
}

__device__ __forceinline__ void wait_async0() {
  asm volatile("s_wait_asynccnt 0" ::: "memory");
}

// ---------------------------------------------------------------------------
// Kernel 1: qkv = x @ w_qkv^T  (fp32 in, f16 out)
// 128x128 tiles, TILE_K = 64, 8 waves, double-buffered LDS.
// ---------------------------------------------------------------------------
__global__ __launch_bounds__(256) void qkv_gemm(const float* __restrict__ x,
                                                const float* __restrict__ w,
                                                _Float16* __restrict__ qkv) {
  constexpr int LD = 72;                       // 64 + 8, row stride 144 B (16-aligned)
  __shared__ __align__(16) _Float16 As[2][128][LD];
  __shared__ __align__(16) _Float16 Bs[2][128][LD];
  const int tid  = threadIdx.x;
  const int lane = tid & 31, wave = tid >> 5;
  const int wm = wave & 3, wn = wave >> 2;
  const int m0 = blockIdx.x * 128;
  const int n0 = blockIdx.y * 128;

  v8f acc[2][4];
#pragma unroll
  for (int i = 0; i < 2; ++i)
#pragma unroll
    for (int j = 0; j < 4; ++j)
#pragma unroll
      for (int e = 0; e < 8; ++e) acc[i][j][e] = 0.0f;

  auto stage = [&](int kt, int buf) {
    const int k0 = kt * 64;
#pragma unroll
    for (int i = 0; i < 8; ++i) {
      const int q  = tid + i * 256;   // 0..2047
      const int r  = q >> 4;          // 0..127
      const int c4 = (q & 15) * 4;    // 0..60
      float4 fa = *reinterpret_cast<const float4*>(&x[(size_t)(m0 + r) * C_ + k0 + c4]);
      As[buf][r][c4 + 0] = (_Float16)fa.x;
      As[buf][r][c4 + 1] = (_Float16)fa.y;
      As[buf][r][c4 + 2] = (_Float16)fa.z;
      As[buf][r][c4 + 3] = (_Float16)fa.w;
      float4 fb = *reinterpret_cast<const float4*>(&w[(size_t)(n0 + r) * C_ + k0 + c4]);
      Bs[buf][r][c4 + 0] = (_Float16)fb.x;
      Bs[buf][r][c4 + 1] = (_Float16)fb.y;
      Bs[buf][r][c4 + 2] = (_Float16)fb.z;
      Bs[buf][r][c4 + 3] = (_Float16)fb.w;
    }
  };

  auto compute = [&](int buf) {
#pragma unroll
    for (int ks = 0; ks < 2; ++ks) {
      v16h a0 = load_afrag(&As[buf][wm * 32][ks * 32], LD, lane);
      v16h a1 = load_afrag(&As[buf][wm * 32 + 16][ks * 32], LD, lane);
#pragma unroll
      for (int ct = 0; ct < 4; ++ct) {
        v16h bf = load_bfrag_nk(&Bs[buf][wn * 64 + ct * 16][ks * 32], LD, lane);
        acc[0][ct] = wmma16(a0, bf, acc[0][ct]);
        acc[1][ct] = wmma16(a1, bf, acc[1][ct]);
      }
    }
  };

  constexpr int NK = C_ / 64;                  // 32 stages
  stage(0, 0);
  __syncthreads();
  for (int kt = 0; kt < NK; ++kt) {
    if (kt + 1 < NK) stage(kt + 1, (kt + 1) & 1);
    compute(kt & 1);
    __syncthreads();
  }

  const int half = lane >> 4, nl = lane & 15;
#pragma unroll
  for (int rt = 0; rt < 2; ++rt)
#pragma unroll
    for (int ct = 0; ct < 4; ++ct)
#pragma unroll
      for (int e = 0; e < 8; ++e) {
        const int row = m0 + wm * 32 + rt * 16 + half * 8 + e;
        const int col = n0 + wn * 64 + ct * 16 + nl;
        qkv[(size_t)row * QKV_F + col] = (_Float16)acc[rt][ct][e];
      }
}

// ---------------------------------------------------------------------------
// Kernel 2: interleaved RoPE on q and k portions of qkv (in place, f16)
// ---------------------------------------------------------------------------
__global__ __launch_bounds__(256) void rope_kernel(_Float16* __restrict__ qkv) {
  const int idx = blockIdx.x * 256 + threadIdx.x;   // 2^24 rotation pairs
  const int p  = idx & 63;                          // pair index within head dim
  const int h  = (idx >> 6) & 15;
  const int qk = (idx >> 10) & 1;                   // 0 = q, 1 = k
  const int bt = idx >> 11;                         // 0..8191
  const int t  = bt & (T_ - 1);
  const size_t base = (size_t)bt * QKV_F + qk * C_ + h * D_ + 2 * p;
  const float xr = (float)qkv[base];
  const float xi = (float)qkv[base + 1];
  const float inv_freq = __powf(10000.0f, -(float)(2 * p) * (1.0f / (float)D_));
  const float ang = (float)t * inv_freq;
  float s, c;
  __sincosf(ang, &s, &c);
  qkv[base]     = (_Float16)(xr * c - xi * s);
  qkv[base + 1] = (_Float16)(xr * s + xi * c);
}

// ---------------------------------------------------------------------------
// Kernel 3: flash attention per (b,h). 8 waves x 16 query rows = 128-row Q tile.
// K/V tiles of 64 keys, double-buffered in LDS and filled with
// global_load_async_to_lds_b128 (ASYNCcnt) one iteration ahead of the WMMAs.
// ---------------------------------------------------------------------------
__global__ __launch_bounds__(256) void flash_attn(const _Float16* __restrict__ qkv,
                                                  _Float16* __restrict__ attn) {
  constexpr int LDQ = D_ + 8;   // 136
  constexpr int LDP = 64 + 8;   // 72
  __shared__ __align__(16) _Float16 Qs[128][LDQ];      // 34816 B
  __shared__ __align__(16) _Float16 Ks[2][64][LDQ];    // 34816 B
  __shared__ __align__(16) _Float16 Vs[2][64][LDQ];    // 34816 B
  __shared__ __align__(16) _Float16 Ps[128][LDP];      // 18432 B

  const int tid  = threadIdx.x;
  const int lane = tid & 31, wave = tid >> 5;
  const int half = lane >> 4, nl = lane & 15;

  const int q0 = blockIdx.x * 128;
  const int bh = blockIdx.y;
  const int b = bh >> 4, h = bh & 15;
  const size_t rowbase = (size_t)b * T_;

  const unsigned long long gbase = (unsigned long long)(uintptr_t)qkv;
  const unsigned long long gq = gbase + (unsigned long long)h * D_ * 2;
  const unsigned long long gk = gq + (unsigned long long)C_ * 2;
  const unsigned long long gv = gq + (unsigned long long)(2 * C_) * 2;

  // ---- async Q tile load (128 x 128 f16) ----
#pragma unroll
  for (int i = 0; i < 8; ++i) {
    const int c  = tid + i * 256;   // 2048 chunks of 8 halves
    const int r  = c >> 4;
    const int d0 = (c & 15) * 8;
    const unsigned voff = (unsigned)(((rowbase + q0 + r) * QKV_F + d0) * 2);
    async_copy_b128((unsigned)(uintptr_t)&Qs[r][d0], voff, gq);
  }

  auto stage_kv = [&](int kt, int buf) {
#pragma unroll
    for (int i = 0; i < 4; ++i) {
      const int c  = tid + i * 256;   // 1024 chunks of 8 halves
      const int r  = c >> 4;          // 0..63
      const int d0 = (c & 15) * 8;
      const unsigned voff =
          (unsigned)(((rowbase + (size_t)kt * 64 + r) * QKV_F + d0) * 2);
      async_copy_b128((unsigned)(uintptr_t)&Ks[buf][r][d0], voff, gk);
      async_copy_b128((unsigned)(uintptr_t)&Vs[buf][r][d0], voff, gv);
    }
  };

  v8f acc_o[8];
  float m_i[8], l_i[8];
#pragma unroll
  for (int ch = 0; ch < 8; ++ch)
#pragma unroll
    for (int e = 0; e < 8; ++e) acc_o[ch][e] = 0.0f;
#pragma unroll
  for (int r = 0; r < 8; ++r) { m_i[r] = -1e30f; l_i[r] = 0.0f; }

  stage_kv(0, 0);
  wait_async0();
  __syncthreads();

  constexpr int NT = T_ / 64;
  for (int kt = 0; kt < NT; ++kt) {
    const int buf = kt & 1;
    if (kt + 1 < NT) stage_kv(kt + 1, buf ^ 1);   // overlap with WMMAs below

    // S = Q K^T : 16 x 64 per wave
    v8f acc_s[4];
#pragma unroll
    for (int ct = 0; ct < 4; ++ct)
#pragma unroll
      for (int e = 0; e < 8; ++e) acc_s[ct][e] = 0.0f;

#pragma unroll
    for (int ks = 0; ks < 4; ++ks) {          // head dim 128 / 32
      v16h aq = load_afrag(&Qs[wave * 16][ks * 32], LDQ, lane);
#pragma unroll
      for (int ct = 0; ct < 4; ++ct) {        // 64 keys / 16
        v16h bk = load_bfrag_nk(&Ks[buf][ct * 16][ks * 32], LDQ, lane);
        acc_s[ct] = wmma16(aq, bk, acc_s[ct]);
      }
    }

    // ---- online softmax (rows live across lane halves: M = r + 8*half) ----
    float mnew[8];
#pragma unroll
    for (int r = 0; r < 8; ++r) {
      float v = -1e30f;
#pragma unroll
      for (int ct = 0; ct < 4; ++ct) v = fmaxf(v, acc_s[ct][r] * SCALE);
      mnew[r] = v;
    }
#pragma unroll
    for (int off = 8; off >= 1; off >>= 1)
#pragma unroll
      for (int r = 0; r < 8; ++r)
        mnew[r] = fmaxf(mnew[r], __shfl_xor(mnew[r], off, 32));

    float rsum[8];
#pragma unroll
    for (int r = 0; r < 8; ++r) {
      const float mn    = fmaxf(m_i[r], mnew[r]);
      const float alpha = __expf(m_i[r] - mn);
      m_i[r] = mn;
      l_i[r] *= alpha;
#pragma unroll
      for (int ch = 0; ch < 8; ++ch) acc_o[ch][r] *= alpha;
      rsum[r] = 0.0f;
    }
#pragma unroll
    for (int ct = 0; ct < 4; ++ct)
#pragma unroll
      for (int r = 0; r < 8; ++r) {
        const float p = __expf(acc_s[ct][r] * SCALE - m_i[r]);
        rsum[r] += p;
        Ps[wave * 16 + r + half * 8][ct * 16 + nl] = (_Float16)p;  // C-layout -> LDS
      }
#pragma unroll
    for (int off = 8; off >= 1; off >>= 1)
#pragma unroll
      for (int r = 0; r < 8; ++r) rsum[r] += __shfl_xor(rsum[r], off, 32);
#pragma unroll
    for (int r = 0; r < 8; ++r) l_i[r] += rsum[r];

    // ---- O += P V (per-wave P region; same-wave LDS ops are in-order) ----
#pragma unroll
    for (int ks = 0; ks < 2; ++ks) {          // 64 keys / 32
      v16h ap = load_afrag(&Ps[wave * 16][ks * 32], LDP, lane);
#pragma unroll
      for (int ch = 0; ch < 8; ++ch) {        // D 128 / 16
        v16h bv = load_bfrag_kn(&Vs[buf][ks * 32][ch * 16], LDQ, lane);
        acc_o[ch] = wmma16(ap, bv, acc_o[ch]);
      }
    }

    wait_async0();     // next tile's async copies have landed
    __syncthreads();   // and everyone is done reading the current buffers
  }

  float inv[8];
#pragma unroll
  for (int r = 0; r < 8; ++r) inv[r] = 1.0f / l_i[r];
#pragma unroll
  for (int ch = 0; ch < 8; ++ch)
#pragma unroll
    for (int r = 0; r < 8; ++r) {
      const int qrow = q0 + wave * 16 + r + half * 8;
      attn[(rowbase + qrow) * (size_t)C_ + h * D_ + ch * 16 + nl] =
          (_Float16)(acc_o[ch][r] * inv[r]);
    }
}

// ---------------------------------------------------------------------------
// Kernel 4: out = attn @ w_proj^T + b_proj  (f16 x fp32-weights -> fp32 out)
// TILE_K = 64; A operand staged with async global->LDS (already f16).
// ---------------------------------------------------------------------------
__global__ __launch_bounds__(256) void proj_gemm(const _Float16* __restrict__ a,
                                                 const float* __restrict__ w,
                                                 const float* __restrict__ bias,
                                                 float* __restrict__ out) {
  constexpr int LD = 72;
  __shared__ __align__(16) _Float16 As[2][128][LD];
  __shared__ __align__(16) _Float16 Bs[2][128][LD];
  const int tid  = threadIdx.x;
  const int lane = tid & 31, wave = tid >> 5;
  const int wm = wave & 3, wn = wave >> 2;
  const int m0 = blockIdx.x * 128;
  const int n0 = blockIdx.y * 128;

  const unsigned long long abase = (unsigned long long)(uintptr_t)a;

  v8f acc[2][4];
#pragma unroll
  for (int i = 0; i < 2; ++i)
#pragma unroll
    for (int j = 0; j < 4; ++j)
#pragma unroll
      for (int e = 0; e < 8; ++e) acc[i][j][e] = 0.0f;

  auto stage = [&](int kt, int buf) {
    const int k0 = kt * 64;
#pragma unroll
    for (int i = 0; i < 4; ++i) {             // A: 128x64 halves via async b128
      const int c  = tid + i * 256;           // 0..1023
      const int r  = c >> 3;                  // 0..127
      const int d0 = (c & 7) * 8;             // 0..56
      const unsigned voff = (unsigned)(((size_t)(m0 + r) * C_ + k0 + d0) * 2);
      async_copy_b128((unsigned)(uintptr_t)&As[buf][r][d0], voff, abase);
    }
#pragma unroll
    for (int i = 0; i < 8; ++i) {             // B: 128x64 floats, convert to f16
      const int q  = tid + i * 256;
      const int r  = q >> 4;
      const int c4 = (q & 15) * 4;
      float4 fb = *reinterpret_cast<const float4*>(&w[(size_t)(n0 + r) * C_ + k0 + c4]);
      Bs[buf][r][c4 + 0] = (_Float16)fb.x;
      Bs[buf][r][c4 + 1] = (_Float16)fb.y;
      Bs[buf][r][c4 + 2] = (_Float16)fb.z;
      Bs[buf][r][c4 + 3] = (_Float16)fb.w;
    }
  };

  auto compute = [&](int buf) {
#pragma unroll
    for (int ks = 0; ks < 2; ++ks) {
      v16h a0 = load_afrag(&As[buf][wm * 32][ks * 32], LD, lane);
      v16h a1 = load_afrag(&As[buf][wm * 32 + 16][ks * 32], LD, lane);
#pragma unroll
      for (int ct = 0; ct < 4; ++ct) {
        v16h bf = load_bfrag_nk(&Bs[buf][wn * 64 + ct * 16][ks * 32], LD, lane);
        acc[0][ct] = wmma16(a0, bf, acc[0][ct]);
        acc[1][ct] = wmma16(a1, bf, acc[1][ct]);
      }
    }
  };

  constexpr int NK = C_ / 64;                  // 32 stages
  stage(0, 0);
  wait_async0();
  __syncthreads();
  for (int kt = 0; kt < NK; ++kt) {
    if (kt + 1 < NK) stage(kt + 1, (kt + 1) & 1);
    compute(kt & 1);
    wait_async0();     // next A tile landed
    __syncthreads();
  }

  const int half = lane >> 4, nl = lane & 15;
#pragma unroll
  for (int rt = 0; rt < 2; ++rt)
#pragma unroll
    for (int ct = 0; ct < 4; ++ct)
#pragma unroll
      for (int e = 0; e < 8; ++e) {
        const int row = m0 + wm * 32 + rt * 16 + half * 8 + e;
        const int col = n0 + wn * 64 + ct * 16 + nl;
        out[(size_t)row * C_ + col] = acc[rt][ct][e] + bias[col];
      }
}

// ---------------------------------------------------------------------------
extern "C" void kernel_launch(void* const* d_in, const int* in_sizes, int n_in,
                              void* d_out, int out_size, void* d_ws, size_t ws_size,
                              hipStream_t stream) {
  (void)in_sizes; (void)n_in; (void)out_size; (void)ws_size;
  const float* x      = (const float*)d_in[0];
  const float* w_qkv  = (const float*)d_in[1];
  const float* w_proj = (const float*)d_in[2];
  const float* b_proj = (const float*)d_in[3];
  float* out = (float*)d_out;

  _Float16* qkv  = reinterpret_cast<_Float16*>(d_ws);                 // B*T*3C f16 (96 MB)
  _Float16* attn = qkv + (size_t)B_ * T_ * QKV_F;                     // B*T*C  f16 (32 MB)

  dim3 g1(B_ * T_ / 128, QKV_F / 128);        // (64, 48)
  qkv_gemm<<<g1, 256, 0, stream>>>(x, w_qkv, qkv);

  const int pairs = B_ * T_ * C_;             // q + k rotation pairs = 2^24
  rope_kernel<<<pairs / 256, 256, 0, stream>>>(qkv);

  dim3 g3(T_ / 128, B_ * H_);                 // (16, 64)
  flash_attn<<<g3, 256, 0, stream>>>(qkv, attn);

  dim3 g4(B_ * T_ / 128, C_ / 128);           // (64, 16)
  proj_gemm<<<g4, 256, 0, stream>>>(attn, w_proj, b_proj, out);
}
